// EIR_Graph_88115549045244
// MI455X (gfx1250) — matrix operand order repaired and checked by reading, not compiled, not run on hardware
//
#include <hip/hip_runtime.h>
#include <hip/hip_bf16.h>
#include <math.h>

// ---------------- CDNA5 WMMA types ----------------
typedef __attribute__((ext_vector_type(16))) __bf16 v16bf;
typedef __attribute__((ext_vector_type(8)))  float  v8f;

#define WMMA_BF16(a, b, c) \
  __builtin_amdgcn_wmma_f32_16x16x32_bf16(false, (a), false, (b), (short)0, (c), false, false)

// Problem dims (fixed by the reference setup)
#define BB  128
#define NN  256
#define MM  12
#define EE  256
#define HH  128
#define DD  3
#define KK  2
#define AF  82
#define AFP 96    // AF padded to multiple of 32
#define BFD 6
#define U2K 160   // (HH+BFD)=134 padded to multiple of 32

// ---------------- device helpers ----------------
__device__ __forceinline__ float dev_lrelu(float x) { return x > 0.f ? x : 0.1f * x; }
__device__ __forceinline__ float dev_sigm(float x)  { return 1.f / (1.f + __expf(-x)); }

__device__ __forceinline__ float apply_act(float x, int act) {
  if (act == 1) return dev_lrelu(x);
  if (act == 2) return tanhf(x);
  if (act == 3) return dev_sigm(x);
  return x;
}

__device__ __forceinline__ v16bf cvt8x2(float4 a0, float4 a1, float4 b0, float4 b1) {
  v16bf r;
  r[0]  = (__bf16)a0.x; r[1]  = (__bf16)a0.y; r[2]  = (__bf16)a0.z; r[3]  = (__bf16)a0.w;
  r[4]  = (__bf16)a1.x; r[5]  = (__bf16)a1.y; r[6]  = (__bf16)a1.z; r[7]  = (__bf16)a1.w;
  r[8]  = (__bf16)b0.x; r[9]  = (__bf16)b0.y; r[10] = (__bf16)b0.z; r[11] = (__bf16)b0.w;
  r[12] = (__bf16)b1.x; r[13] = (__bf16)b1.y; r[14] = (__bf16)b1.z; r[15] = (__bf16)b1.w;
  return r;
}

// A fragment (16x32 bf16) from row-major f32. lane&15 = M row; lane>>4 = K half.
// Per ISA 7.12.2 the per-lane K values form two contiguous 8-float runs
// => 4 aligned float4 loads + 8 v_cvt_pk_bf16_f32, zero branches.
__device__ __forceinline__ v16bf load_A_frag_f32(const float* __restrict__ arow,
                                                 int k0, int half) {
  const float4* p0 = reinterpret_cast<const float4*>(arow + k0 + half * 8);
  const float4* p1 = reinterpret_cast<const float4*>(arow + k0 + 16 + half * 8);
  return cvt8x2(p0[0], p0[1], p1[0], p1[1]);
}

// ---------------- weight packing into B-fragment layout ----------------
// frag index = (kt*ntiles + nt)*32 + lane ; each frag = 16 contiguous bf16
// (lane&15 = N col; elem e<8: K=kt*32+8*half+e, e>=8: K=kt*32+16+8*half+(e-8))
__global__ void pack_B_kernel(const float* __restrict__ W, __bf16* __restrict__ out,
                              int Kv, int Kpad, int N) {
  int batch = blockIdx.y;
  W += (size_t)batch * Kv * N;
  out += (size_t)batch * Kpad * N;
  int i = blockIdx.x * 256 + threadIdx.x;
  if (i >= Kpad * N) return;
  int e = i & 15, lane = (i >> 4) & 31, tile = i >> 9;
  int ntiles = N >> 4;
  int nt = tile % ntiles, kt = tile / ntiles;
  int half = lane >> 4, l15 = lane & 15;
  int k = kt * 32 + ((e < 8) ? (half * 8 + e) : (16 + half * 8 + (e - 8)));
  int n = nt * 16 + l15;
  out[i] = (__bf16)((k < Kv) ? W[(size_t)k * N + n] : 0.f);
}

// ---------------- generic fused WMMA GEMM, 16x64 strip per wave ----------------
// C[M,N] = act( A[M,Kpad] @ Wp + bias[N] + rowext[row>>extshift, N] )
// M % 64 == 0, N % 64 == 0, Kpad % 32 == 0, lda % 4 == 0, A zero-padded in K.
__global__ void gemm_wmma_kernel(const float* __restrict__ A, int lda,
                                 const __bf16* __restrict__ Wp,
                                 const float* __restrict__ bias,
                                 const float* __restrict__ rowext, int extshift,
                                 float* __restrict__ C,
                                 int Kpad, int Ncols, int act) {
  int lane = threadIdx.x;
  int half = lane >> 4, l15 = lane & 15;
  int row0 = (blockIdx.x * 4 + threadIdx.y) * 16;
  int nt0 = blockIdx.y * 4;            // 4 consecutive n-tiles per wave
  int ntiles = Ncols >> 4;
  const float* arow = A + (size_t)(row0 + l15) * lda;
  const v16bf* Wbase = reinterpret_cast<const v16bf*>(Wp) + lane;
  int ktiles = Kpad >> 5;

  v8f acc[4] = {};
  for (int kt = 0; kt < ktiles; ++kt) {
    v16bf af = load_A_frag_f32(arow, kt * 32, half);
    const v16bf* wk = Wbase + ((size_t)kt * ntiles + nt0) * 32;
    acc[0] = WMMA_BF16(af, wk[0],  acc[0]);
    acc[1] = WMMA_BF16(af, wk[32], acc[1]);
    acc[2] = WMMA_BF16(af, wk[64], acc[2]);
    acc[3] = WMMA_BF16(af, wk[96], acc[3]);
  }

#pragma unroll
  for (int t = 0; t < 4; ++t) {
    int n = (nt0 + t) * 16 + l15;
    float bn = bias ? bias[n] : 0.f;
#pragma unroll
    for (int v = 0; v < 8; ++v) {
      int row = row0 + half * 8 + v;
      float x = acc[t][v] + bn;
      if (rowext) x += rowext[(size_t)(row >> extshift) * Ncols + n];
      C[(size_t)row * Ncols + n] = apply_act(x, act);
    }
  }
}

// ---------------- fused attention score ----------------
// a[b,k,n] = sum_g tanh((vf[b]@Wa[k])[n,g] + Wab[k,g]) * sf[b,g]*Wbmm[k,g] + Wbmmb[k]
__global__ void attn_score_kernel(const float* __restrict__ vf,
                                  const float* __restrict__ sf,
                                  const __bf16* __restrict__ WaP,  // packed, k-stride HH*HH
                                  const float* __restrict__ Wab,
                                  const float* __restrict__ Wbmm,
                                  const float* __restrict__ Wbmmb,
                                  float* __restrict__ a_out) {
  int bk = blockIdx.x;
  int b = bk / KK, k = bk % KK;
  int n0 = blockIdx.y * 16;
  int lane = threadIdx.x;
  int half = lane >> 4, l15 = lane & 15;
  const int ntiles = HH >> 4;  // 8

  const float* arow = vf + ((size_t)b * NN + n0 + l15) * HH;
  const v16bf* Wt = reinterpret_cast<const v16bf*>(WaP + (size_t)k * HH * HH) + lane;

  // hoist the whole 16x128 A slab into registers (reused for all 8 g-tiles)
  v16bf afr[HH / 32];
#pragma unroll
  for (int kt = 0; kt < HH / 32; ++kt) afr[kt] = load_A_frag_f32(arow, kt * 32, half);

  float partial[8];
#pragma unroll
  for (int v = 0; v < 8; ++v) partial[v] = 0.f;

#pragma unroll
  for (int gt = 0; gt < ntiles; ++gt) {
    v8f acc = {};
#pragma unroll
    for (int kt = 0; kt < HH / 32; ++kt)
      acc = WMMA_BF16(afr[kt], Wt[((size_t)kt * ntiles + gt) * 32], acc);
    int g = gt * 16 + l15;
    float scale = sf[(size_t)b * HH + g] * Wbmm[(size_t)k * HH + g];
    float bg = Wab[(size_t)k * HH + g];
#pragma unroll
    for (int v = 0; v < 8; ++v) partial[v] += tanhf(acc[v] + bg) * scale;
  }
#pragma unroll
  for (int v = 0; v < 8; ++v) {
    float s = partial[v];
    for (int m = 1; m < 16; m <<= 1) s += __shfl_xor(s, m, 32);
    partial[v] = s;
  }
  if (l15 == 0) {
    float bb = Wbmmb[k];
#pragma unroll
    for (int v = 0; v < 8; ++v)
      a_out[(size_t)bk * NN + n0 + half * 8 + v] = partial[v] + bb;
  }
}

// ---------------- masked softmax over N (LDS reduction) ----------------
__global__ void softmax_mask_kernel(float* __restrict__ a, const float* __restrict__ vmask,
                                    float* __restrict__ ssum) {
  int bk = blockIdx.x;
  int b = bk / KK;
  int n = threadIdx.x;
  __shared__ float red[NN];
  float v = a[(size_t)bk * NN + n];
  red[n] = v; __syncthreads();
  for (int s = NN / 2; s > 0; s >>= 1) { if (n < s) red[n] = fmaxf(red[n], red[n + s]); __syncthreads(); }
  float mx = red[0]; __syncthreads();
  float e = __expf(v - mx) * vmask[(size_t)b * NN + n];
  red[n] = e; __syncthreads();
  for (int s = NN / 2; s > 0; s >>= 1) { if (n < s) red[n] += red[n + s]; __syncthreads(); }
  float denom = red[0] + 1e-6f;
  a[(size_t)bk * NN + n] = e / denom;
  if (n == 0) ssum[bk] = red[0] / denom;
}

// t[b,k,g] = sum_n attn[b,k,n] * vf[b,n,g]
__global__ void attn_apply_kernel(const float* __restrict__ attn, const float* __restrict__ vf,
                                  float* __restrict__ t) {
  int bk = blockIdx.x; int b = bk / KK;
  int g = threadIdx.x;
  float s = 0.f;
  const float* ar = attn + (size_t)bk * NN;
  const float* vb = vf + (size_t)b * NN * HH + g;
  for (int n = 0; n < NN; ++n) s += ar[n] * vb[(size_t)n * HH];
  t[(size_t)bk * HH + g] = s;
}

// m[b,k,h] = t[b,k,:] @ Wmain[k] + ssum[b,k] * Wmain_b[k,h]
__global__ void attn_out_kernel(const float* __restrict__ t, const float* __restrict__ ssum,
                                const float* __restrict__ Wm, const float* __restrict__ Wmb,
                                float* __restrict__ m_out) {
  int bk = blockIdx.x; int k = bk % KK;
  int h = threadIdx.x;
  const float* W = Wm + (size_t)k * HH * HH;
  const float* tr = t + (size_t)bk * HH;
  float s = 0.f;
  for (int g = 0; g < HH; ++g) s += tr[g] * W[(size_t)g * HH + h];
  m_out[(size_t)bk * HH + h] = s + ssum[bk] * Wmb[(size_t)k * HH + h];
}

// ---------------- fused neighbor gather + U2 GEMM + M-reduce ----------------
// One wave per (b,n): gathers the 12 neighbor rows ONCE, computes all 8 h-tiles.
// concat[bn, 128+h] = sum_{r<M} nbs_mask[bn,r]*lrelu((nei[bn,r,:]@U2)[h] + U2b[h])
__global__ void nei_kernel(const float* __restrict__ vf, const float* __restrict__ eini,
                           const int* __restrict__ atom_adj, const int* __restrict__ bond_adj,
                           const float* __restrict__ nbs_mask,
                           const __bf16* __restrict__ U2P, const float* __restrict__ U2b,
                           float* __restrict__ concat) {
  int bn = blockIdx.x;
  int lane = threadIdx.x;
  int half = lane >> 4, l15 = lane & 15;
  const int ntiles = HH >> 4;  // 8

  // invalid rows (r >= M) just read row 0 and are dropped in the epilogue
  int r = (l15 < MM) ? l15 : 0;
  int ai = atom_adj[(size_t)bn * MM + r];
  int bi = bond_adj[(size_t)bn * MM + r];
  const float* vrow = vf + (size_t)ai * HH;
  const float* erow = eini + (size_t)bi * BFD;

  const v16bf* Wt = reinterpret_cast<const v16bf*>(U2P) + lane;

  v8f acc[8] = {};
#pragma unroll
  for (int kt = 0; kt < 4; ++kt) {
    v16bf af = load_A_frag_f32(vrow, kt * 32, half);
#pragma unroll
    for (int ht = 0; ht < 8; ++ht)
      acc[ht] = WMMA_BF16(af, Wt[((size_t)kt * ntiles + ht) * 32], acc[ht]);
  }
  {  // K tail tile (k=128..159): only half==0, elems 0..5 carry bond features
    float mskh = (half == 0) ? 1.f : 0.f;
    v16bf af;
#pragma unroll
    for (int i = 0; i < 16; ++i) af[i] = (__bf16)0.f;
    af[0] = (__bf16)(erow[0] * mskh); af[1] = (__bf16)(erow[1] * mskh);
    af[2] = (__bf16)(erow[2] * mskh); af[3] = (__bf16)(erow[3] * mskh);
    af[4] = (__bf16)(erow[4] * mskh); af[5] = (__bf16)(erow[5] * mskh);
#pragma unroll
    for (int ht = 0; ht < 8; ++ht)
      acc[ht] = WMMA_BF16(af, Wt[((size_t)4 * ntiles + ht) * 32], acc[ht]);
  }

  // per-row neighbor masks (same for all h-tiles)
  float wmask[8];
#pragma unroll
  for (int v = 0; v < 8; ++v) {
    int rr = half * 8 + v;
    wmask[v] = (rr < MM) ? nbs_mask[(size_t)bn * MM + rr] : 0.f;
  }

#pragma unroll
  for (int ht = 0; ht < 8; ++ht) {
    int h = ht * 16 + l15;
    float hb = U2b[h];
    float s = 0.f;
#pragma unroll
    for (int v = 0; v < 8; ++v) s += wmask[v] * dev_lrelu(acc[ht][v] + hb);
    s += __shfl_xor(s, 16, 32);  // combine row halves
    if (half == 0) concat[(size_t)bn * (2 * HH) + HH + h] = s;
  }
}

// ---------------- small elementwise kernels ----------------
__global__ void edge_init_kernel(const int* __restrict__ edge, const float* __restrict__ ibf,
                                 float* __restrict__ eout, int total) {
  int i = blockIdx.x * 256 + threadIdx.x;
  if (i >= total) return;
  int e = i / BFD, c = i % BFD;
  eout[i] = ibf[(size_t)edge[e] * BFD + c];
}

// gather atom features through vertex index into zero-padded (BN x 96) f32
__global__ void gather_pad_atoms(const int* __restrict__ vertex,
                                 const float* __restrict__ init_atom,
                                 float* __restrict__ Apad, int total) {
  int i = blockIdx.x * 256 + threadIdx.x;
  if (i >= total) return;
  int row = i / AFP, c = i % AFP;
  Apad[i] = (c < AF) ? init_atom[(size_t)vertex[row] * AF + c] : 0.f;
}

__global__ void sf_init_kernel(const float* __restrict__ vf, const float* __restrict__ vmask,
                               float* __restrict__ sf) {
  int b = blockIdx.x, h = threadIdx.x;
  float s = 0.f;
  for (int n = 0; n < NN; ++n)
    s += vf[((size_t)b * NN + n) * HH + h] * vmask[(size_t)b * NN + n];
  sf[(size_t)b * HH + h] = s;
}

__global__ void copy_vf_to_concat(const float* __restrict__ vf, float* __restrict__ concat,
                                  int total) {
  int i = blockIdx.x * 256 + threadIdx.x;
  if (i >= total) return;
  int row = i >> 7, c = i & (HH - 1);
  concat[(size_t)row * (2 * HH) + c] = vf[i];
}

__global__ void hidden_main_kernel(const float* __restrict__ z, const float* __restrict__ ms,
                                   const float* __restrict__ s2m, float* __restrict__ hm,
                                   int total) {
  int i = blockIdx.x * 256 + threadIdx.x;
  if (i >= total) return;
  int row = i >> 7, h = i & (HH - 1), b = row >> 8;
  float zz = z[i];
  hm[i] = (1.f - zz) * ms[i] + zz * s2m[(size_t)b * HH + h];
}

__global__ void hidden_super_kernel(const float* __restrict__ z, const float* __restrict__ ss,
                                    const float* __restrict__ m2s, float* __restrict__ hs,
                                    int total) {
  int i = blockIdx.x * 256 + threadIdx.x;
  if (i >= total) return;
  float zz = z[i];
  hs[i] = (1.f - zz) * ss[i] + zz * m2s[i];
}

__global__ void gru_gate_kernel(const float* __restrict__ gi, const float* __restrict__ gh,
                                const float* __restrict__ hprev, float* __restrict__ hout,
                                int total) {
  int i = blockIdx.x * 256 + threadIdx.x;
  if (i >= total) return;
  int row = i >> 7, h = i & (HH - 1);
  const float* gir = gi + (size_t)row * 3 * HH;
  const float* ghr = gh + (size_t)row * 3 * HH;
  float r = dev_sigm(gir[h] + ghr[h]);
  float z = dev_sigm(gir[HH + h] + ghr[HH + h]);
  float n = tanhf(gir[2 * HH + h] + r * ghr[2 * HH + h]);
  hout[i] = (1.f - z) * n + z * hprev[i];
}

// ---------------- host orchestration ----------------
static inline void launch_gemm(const float* A, int lda, const __bf16* Wp,
                               const float* bias, const float* rowext, int extshift,
                               float* C, int M, int Kpad, int N, int act, hipStream_t s) {
  dim3 grid(M / 64, N / 64);
  dim3 blk(32, 4);
  gemm_wmma_kernel<<<grid, blk, 0, s>>>(A, lda, Wp, bias, rowext, extshift, C, Kpad, N, act);
}

static inline void launch_pack(const float* W, __bf16* out, int Kv, int Kpad, int N,
                               int batch, hipStream_t s) {
  dim3 grid((Kpad * N + 255) / 256, batch);
  pack_B_kernel<<<grid, 256, 0, s>>>(W, out, Kv, Kpad, N);
}

extern "C" void kernel_launch(void* const* d_in, const int* in_sizes, int n_in,
                              void* d_out, int out_size, void* d_ws, size_t ws_size,
                              hipStream_t stream) {
  (void)in_sizes; (void)n_in; (void)out_size; (void)ws_size;
  const float* vertex_mask = (const float*)d_in[1];
  const int*   vertex      = (const int*)d_in[2];
  const int*   edge        = (const int*)d_in[3];
  const int*   atom_adj    = (const int*)d_in[4];
  const int*   bond_adj    = (const int*)d_in[5];
  const float* nbs_mask    = (const float*)d_in[6];
  const float* init_atom   = (const float*)d_in[7];
  const float* init_bond   = (const float*)d_in[8];
  const float* emb_w  = (const float*)d_in[9];
  const float* emb_b  = (const float*)d_in[10];
  const float* Wa_w   = (const float*)d_in[11];
  const float* Wa_b   = (const float*)d_in[12];
  const float* Wbmm_w = (const float*)d_in[13];
  const float* Wbmm_b = (const float*)d_in[14];
  const float* Wm_w   = (const float*)d_in[15];
  const float* Wm_b   = (const float*)d_in[16];
  const float* Wm2s_w = (const float*)d_in[17];
  const float* Wm2s_b = (const float*)d_in[18];
  const float* Ws2m_w = (const float*)d_in[19];
  const float* Ws2m_b = (const float*)d_in[20];
  const float* Wsp_w  = (const float*)d_in[21];
  const float* Wsp_b  = (const float*)d_in[22];
  const float* Wzm1_w = (const float*)d_in[23];
  const float* Wzm1_b = (const float*)d_in[24];
  const float* Wzm2_w = (const float*)d_in[25];
  const float* Wzm2_b = (const float*)d_in[26];
  const float* Wzs1_w = (const float*)d_in[27];
  const float* Wzs1_b = (const float*)d_in[28];
  const float* Wzs2_w = (const float*)d_in[29];
  const float* Wzs2_b = (const float*)d_in[30];
  const float* U2_w   = (const float*)d_in[31];
  const float* U2_b   = (const float*)d_in[32];
  const float* U1_w   = (const float*)d_in[33];
  const float* U1_b   = (const float*)d_in[34];
  const float* gm_wih = (const float*)d_in[35];
  const float* gm_whh = (const float*)d_in[36];
  const float* gm_bih = (const float*)d_in[37];
  const float* gm_bhh = (const float*)d_in[38];
  const float* gs_wih = (const float*)d_in[39];
  const float* gs_whh = (const float*)d_in[40];
  const float* gs_bih = (const float*)d_in[41];
  const float* gs_bhh = (const float*)d_in[42];

  const int BN = BB * NN;

  // 32-byte aligned bump allocator on d_ws
  size_t off = 0;
  float* ws = (float*)d_ws;
  auto alloc = [&](size_t n) { float* p = ws + off; off += (n + 7) & ~(size_t)7; return p; };

  float* Apad = alloc((size_t)BN * AFP);
  float* vfA  = alloc((size_t)BN * HH);
  float* vfB  = alloc((size_t)BN * HH);
  float* sfA  = alloc((size_t)BB * HH);
  float* sfB  = alloc((size_t)BB * HH);
  float* eini = alloc((size_t)BB * EE * BFD);
  float* ascr = alloc((size_t)BB * KK * NN);
  float* ssum = alloc((size_t)BB * KK);
  float* tbuf = alloc((size_t)BB * KK * HH);
  float* mbuf = alloc((size_t)BB * KK * HH);
  float* m2s  = alloc((size_t)BB * HH);
  float* cat  = alloc((size_t)BN * 2 * HH);
  float* msf  = alloc((size_t)BN * HH);
  float* zmn  = alloc((size_t)BN * HH);
  float* hm   = alloc((size_t)BN * HH);
  float* giM  = alloc((size_t)BN * 3 * HH);
  float* ghM  = alloc((size_t)BN * 3 * HH);
  float* s2m  = alloc((size_t)BB * HH);
  float* ssb  = alloc((size_t)BB * HH);
  float* szm  = alloc((size_t)BB * HH);
  float* tmpS = alloc((size_t)BB * HH);
  float* zsb  = alloc((size_t)BB * HH);
  float* hs   = alloc((size_t)BB * HH);
  float* giS  = alloc((size_t)BB * 3 * HH);
  float* ghS  = alloc((size_t)BB * 3 * HH);

  // packed bf16 weights (counts are bf16 elements; alloc takes floats)
  auto balloc = [&](size_t nbf) { return (__bf16*)alloc((nbf + 1) / 2); };
  __bf16* embP  = balloc((size_t)AFP * HH);
  __bf16* WaP   = balloc((size_t)DD * KK * HH * HH);
  __bf16* Wm2sP = balloc((size_t)DD * 2 * HH * HH);
  __bf16* Ws2mP = balloc((size_t)DD * HH * HH);
  __bf16* WspP  = balloc((size_t)DD * HH * HH);
  __bf16* Wzm1P = balloc((size_t)DD * HH * HH);
  __bf16* Wzm2P = balloc((size_t)DD * HH * HH);
  __bf16* Wzs1P = balloc((size_t)DD * HH * HH);
  __bf16* Wzs2P = balloc((size_t)DD * HH * HH);
  __bf16* U2P   = balloc((size_t)DD * U2K * HH);
  __bf16* U1P   = balloc((size_t)DD * 2 * HH * HH);
  __bf16* gmIhP = balloc((size_t)HH * 3 * HH);
  __bf16* gmHhP = balloc((size_t)HH * 3 * HH);
  __bf16* gsIhP = balloc((size_t)HH * 3 * HH);
  __bf16* gsHhP = balloc((size_t)HH * 3 * HH);

  // ---- one-time packing ----
  launch_pack(emb_w,  embP,  AF, AFP, HH, 1, stream);
  launch_pack(Wa_w,   WaP,   HH, HH, HH, DD * KK, stream);
  launch_pack(Wm2s_w, Wm2sP, 2 * HH, 2 * HH, HH, DD, stream);
  launch_pack(Ws2m_w, Ws2mP, HH, HH, HH, DD, stream);
  launch_pack(Wsp_w,  WspP,  HH, HH, HH, DD, stream);
  launch_pack(Wzm1_w, Wzm1P, HH, HH, HH, DD, stream);
  launch_pack(Wzm2_w, Wzm2P, HH, HH, HH, DD, stream);
  launch_pack(Wzs1_w, Wzs1P, HH, HH, HH, DD, stream);
  launch_pack(Wzs2_w, Wzs2P, HH, HH, HH, DD, stream);
  launch_pack(U2_w,   U2P,   HH + BFD, U2K, HH, DD, stream);
  launch_pack(U1_w,   U1P,   2 * HH, 2 * HH, HH, DD, stream);
  launch_pack(gm_wih, gmIhP, HH, HH, 3 * HH, 1, stream);
  launch_pack(gm_whh, gmHhP, HH, HH, 3 * HH, 1, stream);
  launch_pack(gs_wih, gsIhP, HH, HH, 3 * HH, 1, stream);
  launch_pack(gs_whh, gsHhP, HH, HH, 3 * HH, 1, stream);

  // ---- preprocessing ----
  edge_init_kernel<<<(BB * EE * BFD + 255) / 256, 256, 0, stream>>>(edge, init_bond, eini,
                                                                    BB * EE * BFD);
  gather_pad_atoms<<<(BN * AFP + 255) / 256, 256, 0, stream>>>(vertex, init_atom, Apad,
                                                               BN * AFP);
  // vf = lrelu(atoms @ emb_w + emb_b)
  launch_gemm(Apad, AFP, embP, emb_b, nullptr, 0, vfA, BN, AFP, HH, 1, stream);
  sf_init_kernel<<<BB, HH, 0, stream>>>(vfA, vertex_mask, sfA);

  float* vf_cur = vfA; float* vf_nxt = vfB;
  float* sf_cur = sfA; float* sf_nxt = sfB;

  for (int d = 0; d < DD; ++d) {
    // attention scores (fused tanh-proj + sf/Wbmm contraction)
    attn_score_kernel<<<dim3(BB * KK, NN / 16), 32, 0, stream>>>(
        vf_cur, sf_cur, WaP + (size_t)d * KK * HH * HH,
        Wa_b + (size_t)d * KK * HH, Wbmm_w + (size_t)d * KK * HH,
        Wbmm_b + (size_t)d * KK, ascr);
    softmax_mask_kernel<<<BB * KK, NN, 0, stream>>>(ascr, vertex_mask, ssum);
    attn_apply_kernel<<<BB * KK, HH, 0, stream>>>(ascr, vf_cur, tbuf);
    attn_out_kernel<<<BB * KK, HH, 0, stream>>>(
        tbuf, ssum, Wm_w + (size_t)d * KK * HH * HH, Wm_b + (size_t)d * KK * HH, mbuf);
    // main_to_super = tanh(m.reshape(B, K*H) @ Wm2s + b)
    launch_gemm(mbuf, KK * HH, Wm2sP + (size_t)d * 2 * HH * HH,
                Wm2s_b + (size_t)d * HH, nullptr, 0, m2s, BB, 2 * HH, HH, 2, stream);

    // neighbor gather + U2 + reduce -> concat[:,128:256]; vf -> concat[:,0:128]
    copy_vf_to_concat<<<(BN * HH + 255) / 256, 256, 0, stream>>>(vf_cur, cat, BN * HH);
    nei_kernel<<<BN, 32, 0, stream>>>(
        vf_cur, eini, atom_adj, bond_adj, nbs_mask,
        U2P + (size_t)d * U2K * HH, U2_b + (size_t)d * HH, cat);
    // main_self = lrelu(concat @ U1 + b)
    launch_gemm(cat, 2 * HH, U1P + (size_t)d * 2 * HH * HH,
                U1_b + (size_t)d * HH, nullptr, 0, msf, BN, 2 * HH, HH, 1, stream);

    // super projections
    launch_gemm(sf_cur, HH, Ws2mP + (size_t)d * HH * HH,
                Ws2m_b + (size_t)d * HH, nullptr, 0, s2m, BB, HH, HH, 2, stream);
    launch_gemm(sf_cur, HH, WspP + (size_t)d * HH * HH,
                Wsp_b + (size_t)d * HH, nullptr, 0, ssb, BB, HH, HH, 2, stream);
    launch_gemm(s2m, HH, Wzm2P + (size_t)d * HH * HH,
                Wzm2_b + (size_t)d * HH, nullptr, 0, szm, BB, HH, HH, 0, stream);
    // z_main = sigmoid(main_self @ Wzm1 + b + szm[row>>8])   (256 rows per batch)
    launch_gemm(msf, HH, Wzm1P + (size_t)d * HH * HH,
                Wzm1_b + (size_t)d * HH, szm, 8, zmn, BN, HH, HH, 3, stream);
    hidden_main_kernel<<<(BN * HH + 255) / 256, 256, 0, stream>>>(zmn, msf, s2m, hm, BN * HH);

    // main GRU
    launch_gemm(hm, HH, gmIhP, gm_bih, nullptr, 0, giM, BN, HH, 3 * HH, 0, stream);
    launch_gemm(vf_cur, HH, gmHhP, gm_bhh, nullptr, 0, ghM, BN, HH, 3 * HH, 0, stream);
    gru_gate_kernel<<<(BN * HH + 255) / 256, 256, 0, stream>>>(giM, ghM, vf_cur, vf_nxt,
                                                               BN * HH);

    // super gate + GRU
    launch_gemm(ssb, HH, Wzs1P + (size_t)d * HH * HH,
                Wzs1_b + (size_t)d * HH, nullptr, 0, tmpS, BB, HH, HH, 0, stream);
    launch_gemm(m2s, HH, Wzs2P + (size_t)d * HH * HH,
                Wzs2_b + (size_t)d * HH, tmpS, 0, zsb, BB, HH, HH, 3, stream);
    hidden_super_kernel<<<(BB * HH + 255) / 256, 256, 0, stream>>>(zsb, ssb, m2s, hs, BB * HH);
    launch_gemm(hs, HH, gsIhP, gs_bih, nullptr, 0, giS, BB, HH, 3 * HH, 0, stream);
    launch_gemm(sf_cur, HH, gsHhP, gs_bhh, nullptr, 0, ghS, BB, HH, 3 * HH, 0, stream);
    gru_gate_kernel<<<(BB * HH + 255) / 256, 256, 0, stream>>>(giS, ghS, sf_cur, sf_nxt,
                                                               BB * HH);

    float* t1 = vf_cur; vf_cur = vf_nxt; vf_nxt = t1;
    float* t2 = sf_cur; sf_cur = sf_nxt; sf_nxt = t2;
  }

  float* out = (float*)d_out;
  hipMemcpyAsync(out, vf_cur, (size_t)BN * HH * sizeof(float), hipMemcpyDeviceToDevice, stream);
  hipMemcpyAsync(out + (size_t)BN * HH, sf_cur, (size_t)BB * HH * sizeof(float),
                 hipMemcpyDeviceToDevice, stream);
}
